// PIDSparseAttentionV3_627065225879
// MI455X (gfx1250) — matrix operand order repaired and checked by reading, not compiled, not run on hardware
//
#include <hip/hip_runtime.h>
#include <hip/hip_bf16.h>

// ---------------------------------------------------------------------------
// PID sparse attention, MI455X (gfx1250, wave32, WMMA, async-LDS loads).
//   1) chunked EMA scan (integral)
//   2) per-row stats (14 dot-products)
//   3) gates / kick coefficients
//   4) weight concat+bf16, A-matrix build (gates + deriv kick + EMA fixup)
//   5) 3x NT-GEMM (4096x1024x3072) bf16 WMMA, double-buffered async-LDS
//   6) dilated sparse attention, wave32/query
//   7) NT-GEMM (4096x1024x1024) bf16 WMMA -> fp32 d_out
// ---------------------------------------------------------------------------

typedef __bf16 bf16;
typedef __attribute__((ext_vector_type(16))) __bf16 bf16x16;
typedef __attribute__((ext_vector_type(8)))  __bf16 bf16x8;
typedef __attribute__((ext_vector_type(8)))  float  f32x8;

constexpr int B_ = 2, T_ = 2048, D_ = 1024, H_ = 16, AMAX = 15;
constexpr int BT = B_ * T_;
constexpr int CL = 64, NC = T_ / CL;      // scan chunk length / count
constexpr float DECAY = 0.95f;

// ------------------------------ scan ---------------------------------------
__global__ void k_scan_local(const float* __restrict__ x,
                             float* __restrict__ z, float* __restrict__ zlast) {
  int tid = blockIdx.x * blockDim.x + threadIdx.x;
  if (tid >= NC * B_ * D_) return;
  int d = tid % D_;
  int rest = tid / D_;
  int b = rest % B_;
  int c = rest / B_;
  int t0 = c * CL;
  float carry = 0.f;
  for (int i = 0; i < CL; ++i) {
    int t = t0 + i;
    float xv = x[((size_t)(b * T_ + t)) * D_ + d];
    carry = (c == 0 && i == 0) ? xv : DECAY * carry + (1.f - DECAY) * xv;
    z[((size_t)(b * T_ + t)) * D_ + d] = carry;
  }
  zlast[tid] = carry;   // layout [c][b][d]
}

__global__ void k_scan_carry(const float* __restrict__ zlast,
                             float* __restrict__ carryin) {
  int tid = blockIdx.x * blockDim.x + threadIdx.x;
  if (tid >= B_ * D_) return;
  int d = tid % D_;
  int b = tid / D_;
  float a64 = __powf(DECAY, (float)CL);
  float y = 0.f;
  for (int c = 0; c < NC; ++c) {
    size_t idx = ((size_t)(c * B_ + b)) * D_ + d;
    carryin[idx] = y;                    // carry entering chunk c
    y = zlast[idx] + a64 * y;
  }
}

// ------------------------------ stats --------------------------------------
__global__ __launch_bounds__(256)
void k_stats(const float* __restrict__ x, const float* __restrict__ kick_w,
             const float* __restrict__ gate_w, float* __restrict__ stats) {
  int row = blockIdx.x;                  // b*T + t
  int t = row % T_;
  int tid = threadIdx.x;
  float acc[14];
#pragma unroll
  for (int i = 0; i < 14; ++i) acc[i] = 0.f;
  for (int d = tid; d < D_; d += 256) {
    float xv = x[(size_t)row * D_ + d];
    float xp = (t > 0) ? x[(size_t)(row - 1) * D_ + d] : 0.f;
    acc[0] += xv * xv;
    acc[1] += xv * xp;
#pragma unroll
    for (int p = 0; p < 3; ++p) acc[2 + p] += xv * kick_w[p * D_ + d];
#pragma unroll
    for (int pj = 0; pj < 9; ++pj) acc[5 + pj] += xv * gate_w[(size_t)pj * D_ + d];
  }
  __shared__ float red[256];
  for (int i = 0; i < 14; ++i) {
    red[tid] = acc[i];
    __syncthreads();
    for (int s = 128; s > 0; s >>= 1) {
      if (tid < s) red[tid] += red[tid + s];
      __syncthreads();
    }
    if (tid == 0) stats[(size_t)row * 14 + i] = red[0];
    __syncthreads();
  }
}

__global__ void k_gates(const float* __restrict__ stats,
                        const float* __restrict__ gate_b,
                        const float* __restrict__ kick_b,
                        float* __restrict__ g, float* __restrict__ kc) {
  int row = blockIdx.x * blockDim.x + threadIdx.x;
  if (row >= BT) return;
  int t = row % T_;
  const float* s = stats + (size_t)row * 14;
  float nrm = fmaxf(sqrtf(s[0]), 1e-12f);
  float cosv = 0.f;
  if (t > 0) {
    float np = fmaxf(sqrtf(stats[(size_t)(row - 1) * 14]), 1e-12f);
    cosv = s[1] / (nrm * np);
  }
  float stag = (cosv > 0.95f) ? 1.f : 0.f;
#pragma unroll
  for (int p = 0; p < 3; ++p) {
    float ks = 1.f / (1.f + __expf(-(s[2 + p] + kick_b[p])));
    kc[row * 3 + p] = stag * ks;
    float l0 = s[5 + p * 3 + 0] + gate_b[p * 3 + 0];
    float l1 = s[5 + p * 3 + 1] + gate_b[p * 3 + 1];
    float l2 = s[5 + p * 3 + 2] + gate_b[p * 3 + 2];
    float m = fmaxf(l0, fmaxf(l1, l2));
    float e0 = __expf(l0 - m), e1 = __expf(l1 - m), e2 = __expf(l2 - m);
    float inv = 1.f / (e0 + e1 + e2);
    float g0 = fminf(e0 * inv, 0.6f);
    float g1 = fminf(e1 * inv, 0.6f);
    float g2 = fminf(e2 * inv, 0.6f);
    g1 = fmaxf(g1, 0.25f);
    float gi = 1.f / (g0 + g1 + g2);
    g[((size_t)row * 3 + p) * 3 + 0] = g0 * gi;
    g[((size_t)row * 3 + p) * 3 + 1] = g1 * gi;
    g[((size_t)row * 3 + p) * 3 + 2] = g2 * gi;
  }
}

// ----------------------- weight / input conversion -------------------------
__global__ void k_wcat(const float* __restrict__ Wp, const float* __restrict__ Wi,
                       const float* __restrict__ Wd, bf16* __restrict__ Wc) {
  size_t id = (size_t)blockIdx.x * blockDim.x + threadIdx.x;
  if (id >= 3ull * 1024 * 3072) return;
  int k = (int)(id % 3072);
  size_t pn = id / 3072;
  int n = (int)(pn % 1024);
  int p = (int)(pn / 1024);
  const float* src = (k < 1024) ? Wp : ((k < 2048) ? Wi : Wd);
  int kk = k & 1023;
  Wc[id] = (bf16)src[((size_t)p * 1024 + n) * 1024 + kk];
}

__global__ void k_f32_to_bf16(const float* __restrict__ in, bf16* __restrict__ out,
                              size_t n) {
  size_t id = (size_t)blockIdx.x * blockDim.x + threadIdx.x;
  if (id < n) out[id] = (bf16)in[id];
}

__global__ void k_buildA(const float* __restrict__ x, const float* __restrict__ z,
                         const float* __restrict__ carryin,
                         const float* __restrict__ g, const float* __restrict__ kc,
                         const float* __restrict__ kick_v, bf16* __restrict__ A) {
  size_t id = (size_t)blockIdx.x * blockDim.x + threadIdx.x;
  if (id >= (size_t)BT * D_) return;
  int d = (int)(id % D_);
  int row = (int)(id / D_);
  int t = row % T_;
  int b = row / T_;
  float xv = x[id];
  float dv = (t > 0) ? xv - x[id - D_] : 0.f;
  int i = t & (CL - 1);
  int c = t >> 6;
  float ci = carryin[((size_t)(c * B_ + b)) * D_ + d];
  float I = z[id] + __powf(DECAY, (float)(i + 1)) * ci;   // EMA fixup
#pragma unroll
  for (int p = 0; p < 3; ++p) {
    const float* gp = g + ((size_t)row * 3 + p) * 3;
    float dvp = dv + kc[row * 3 + p] * kick_v[p * D_ + d];
    bf16* Ap = A + (size_t)p * BT * 3072 + (size_t)row * 3072;
    Ap[d] = (bf16)(gp[0] * xv);
    Ap[1024 + d] = (bf16)(gp[1] * I);
    Ap[2048 + d] = (bf16)(gp[2] * dvp);
  }
}

// ---------------------- async global->LDS helpers --------------------------
// GLOBAL_LOAD_ASYNC_TO_LDS_B128 (GV mode), tracked by ASYNCcnt (ISA 10.x/15.18.3).
// LDS byte offset = low 32 bits of the generic pointer (flat->LDS truncation).
__device__ inline unsigned lds_addr32(const void* p) {
  return (unsigned)(unsigned long long)(uintptr_t)p;
}
__device__ inline void async_load_b128(unsigned lds, const void* gptr) {
  asm volatile("global_load_async_to_lds_b128 %0, %1, off"
               :: "v"(lds), "v"((unsigned long long)(uintptr_t)gptr)
               : "memory");
}
#define WAIT_ASYNCCNT(imm) asm volatile("s_wait_asynccnt " #imm ::: "memory")

// ------------------------------ WMMA GEMM ----------------------------------
// C[M,N] = A[M,K] * W[N,K]^T  (NT, bf16 in, f32 accumulate).
// 256 thr = 8 waves; block tile 128x128x32; wave tile 32x64 = 2x4 WMMA frags.
// Double-buffered LDS fed by global_load_async_to_lds_b128 (4 per thr/tile).
template <int OUT_BF16>
__global__ __launch_bounds__(256)
void k_gemm_bf16(const bf16* __restrict__ A, const bf16* __restrict__ W,
                 float* __restrict__ Cf, bf16* __restrict__ Cb,
                 int M, int N, int K,
                 long long sA, long long sW, long long sC) {
  constexpr int TM = 128, TN = 128, TK = 32, LDT = 40;  // pad 32->40 (bank-safe)
  __shared__ __align__(16) bf16 As[2][TM * LDT];
  __shared__ __align__(16) bf16 Bs[2][TN * LDT];
  const int tid = threadIdx.x;
  const int wave = tid >> 5, lane = tid & 31;
  const int wm = wave & 3, wn = wave >> 2;
  const int m0 = wm * 32, n0 = wn * 64;
  const int bn = blockIdx.x * TN, bm = blockIdx.y * TM;
  const bf16* Ab = A + (size_t)blockIdx.z * sA + (size_t)bm * K;
  const bf16* Wb = W + (size_t)blockIdx.z * sW + (size_t)bn * K;
  const int half = lane >> 4, l15 = lane & 15;

  // per-thread tile-load slots: 2 A segments + 2 B segments (16B each)
  const int rA0 = tid >> 2,            cA0 = (tid & 3) * 8;
  const int rA1 = (tid + 256) >> 2,    cA1 = cA0;            // seg tid+256
  const int rB0 = rA0, cB0 = cA0, rB1 = rA1, cB1 = cA1;

  auto issue_tile = [&](int k0, int buf) {
    async_load_b128(lds_addr32(&As[buf][rA0 * LDT + cA0]),
                    Ab + (size_t)rA0 * K + k0 + cA0);
    async_load_b128(lds_addr32(&As[buf][rA1 * LDT + cA1]),
                    Ab + (size_t)rA1 * K + k0 + cA1);
    async_load_b128(lds_addr32(&Bs[buf][rB0 * LDT + cB0]),
                    Wb + (size_t)rB0 * K + k0 + cB0);
    async_load_b128(lds_addr32(&Bs[buf][rB1 * LDT + cB1]),
                    Wb + (size_t)rB1 * K + k0 + cB1);
  };

  f32x8 acc[2][4] = {};
  const int nsteps = K / TK;

  issue_tile(0, 0);
  for (int t = 0; t < nsteps; ++t) {
    const int buf = t & 1;
    if (t + 1 < nsteps) {
      issue_tile((t + 1) * TK, buf ^ 1);
      WAIT_ASYNCCNT(0x4);   // tile t's 4 transfers done; tile t+1 in flight
    } else {
      WAIT_ASYNCCNT(0x0);
    }
    __syncthreads();        // all waves' tile t resident in LDS

    // A frag (16x32): lanes 0-15 -> K {0..7,16..23}; lanes 16-31 -> {8..15,24..31}
    bf16x16 aF[2], bF[4];
#pragma unroll
    for (int mi = 0; mi < 2; ++mi) {
      const bf16* src = &As[buf][(m0 + mi * 16 + l15) * LDT];
      bf16x8 lo = *reinterpret_cast<const bf16x8*>(src + half * 8);
      bf16x8 hi = *reinterpret_cast<const bf16x8*>(src + 16 + half * 8);
      aF[mi] = __builtin_shufflevector(lo, hi, 0, 1, 2, 3, 4, 5, 6, 7,
                                       8, 9, 10, 11, 12, 13, 14, 15);
    }
    // B frag (32x16): lanes 0-15 -> K {0..7,16..23}; lanes 16-31 -> {8..15,24..31}
#pragma unroll
    for (int ni = 0; ni < 4; ++ni) {
      const bf16* src = &Bs[buf][(n0 + ni * 16 + l15) * LDT];
      bf16x8 lo = *reinterpret_cast<const bf16x8*>(src + half * 8);
      bf16x8 hi = *reinterpret_cast<const bf16x8*>(src + 16 + half * 8);
      bF[ni] = __builtin_shufflevector(lo, hi, 0, 1, 2, 3, 4, 5, 6, 7,
                                       8, 9, 10, 11, 12, 13, 14, 15);
    }
#pragma unroll
    for (int mi = 0; mi < 2; ++mi)
#pragma unroll
      for (int ni = 0; ni < 4; ++ni)
        acc[mi][ni] = __builtin_amdgcn_wmma_f32_16x16x32_bf16(
            false, aF[mi], false, bF[ni], (short)0, acc[mi][ni], false, false);
    __syncthreads();        // readers done before buf is refilled (t+2)
  }

  // C/D layout: VGPR r -> M = 8*(lane>=16) + r ; N = lane&15
#pragma unroll
  for (int mi = 0; mi < 2; ++mi)
#pragma unroll
    for (int ni = 0; ni < 4; ++ni) {
      int col = bn + n0 + ni * 16 + l15;
      int rowb = bm + m0 + mi * 16 + half * 8;
#pragma unroll
      for (int r = 0; r < 8; ++r) {
        float v = acc[mi][ni][r];
        size_t idx = (size_t)blockIdx.z * sC + (size_t)(rowb + r) * N + col;
        if constexpr (OUT_BF16) Cb[idx] = (bf16)v;
        else                    Cf[idx] = v;
      }
    }
}

// ------------------------------ attention ----------------------------------
__device__ inline float wave_sum32(float v) {
#pragma unroll
  for (int o = 16; o >= 1; o >>= 1) v += __shfl_xor(v, o, 32);
  return v;
}

__global__ __launch_bounds__(256)
void k_attn(const bf16* __restrict__ qkv, const int* __restrict__ positions,
            bf16* __restrict__ out) {
  int qi = blockIdx.x * 8 + (threadIdx.x >> 5);
  if (qi >= B_ * H_ * T_) return;
  int lane = threadIdx.x & 31;
  int t = qi % T_;
  int h = (qi / T_) % H_;
  int b = qi / (T_ * H_);
  const size_t SP = (size_t)BT * 1024;
  const bf16* Q = qkv;
  const bf16* Kk = qkv + SP;
  const bf16* V = qkv + 2 * SP;
  size_t qbase = ((size_t)(b * T_ + t)) * 1024 + h * 64;
  int d0 = lane, d1 = lane + 32;
  float q0 = (float)Q[qbase + d0], q1 = (float)Q[qbase + d1];

  int na = 1 + min(3, t);
  for (int k = 2; (1 << k) <= t; ++k) ++na;   // dilated positions count
  const int* pr = positions + t * AMAX;

  float s[AMAX];
  for (int a = 0; a < na; ++a) {
    int p = pr[a];
    size_t kb = ((size_t)(b * T_ + p)) * 1024 + h * 64;
    float partial = q0 * (float)Kk[kb + d0] + q1 * (float)Kk[kb + d1];
    s[a] = wave_sum32(partial) * 0.125f;      // 1/sqrt(64)
  }
  float m = -1e30f;
  for (int a = 0; a < na; ++a) m = fmaxf(m, s[a]);
  float sum = 0.f, o0 = 0.f, o1 = 0.f;
  for (int a = 0; a < na; ++a) {
    int p = pr[a];
    size_t vb = ((size_t)(b * T_ + p)) * 1024 + h * 64;
    float w = __expf(s[a] - m);
    sum += w;
    o0 += w * (float)V[vb + d0];
    o1 += w * (float)V[vb + d1];
  }
  float inv = 1.f / sum;
  out[qbase + d0] = (bf16)(o0 * inv);
  out[qbase + d1] = (bf16)(o1 * inv);
}

// ------------------------------ launch -------------------------------------
extern "C" void kernel_launch(void* const* d_in, const int* in_sizes, int n_in,
                              void* d_out, int out_size, void* d_ws, size_t ws_size,
                              hipStream_t stream) {
  const float* x       = (const float*)d_in[0];
  const float* Wp      = (const float*)d_in[1];
  const float* Wi      = (const float*)d_in[2];
  const float* Wd      = (const float*)d_in[3];
  const float* gate_w  = (const float*)d_in[4];
  const float* gate_b  = (const float*)d_in[5];
  const float* kick_v  = (const float*)d_in[6];
  const float* kick_w  = (const float*)d_in[7];
  const float* kick_b  = (const float*)d_in[8];
  const float* o_w     = (const float*)d_in[9];
  const int*   positions = (const int*)d_in[10];
  float* outp = (float*)d_out;

  // workspace bump allocator (total ~148 MB)
  size_t off = 0;
  auto alloc = [&](size_t bytes) -> char* {
    char* p = (char*)d_ws + off;
    off = (off + bytes + 255) & ~(size_t)255;
    return p;
  };
  float* z       = (float*)alloc((size_t)BT * D_ * 4);
  float* zlast   = (float*)alloc((size_t)NC * B_ * D_ * 4);
  float* carryin = (float*)alloc((size_t)NC * B_ * D_ * 4);
  float* stats   = (float*)alloc((size_t)BT * 14 * 4);
  float* g       = (float*)alloc((size_t)BT * 9 * 4);
  float* kc      = (float*)alloc((size_t)BT * 3 * 4);
  bf16*  Wcat    = (bf16*)alloc(3ull * 1024 * 3072 * 2);
  bf16*  owb     = (bf16*)alloc(1024ull * 1024 * 2);
  bf16*  Abuf    = (bf16*)alloc(3ull * BT * 3072 * 2);
  bf16*  qkvb    = (bf16*)alloc(3ull * BT * 1024 * 2);
  bf16*  attnb   = (bf16*)alloc((size_t)BT * 1024 * 2);
  (void)ws_size; (void)n_in; (void)in_sizes; (void)out_size;

  // 1) EMA scan
  k_scan_local<<<(NC * B_ * D_ + 255) / 256, 256, 0, stream>>>(x, z, zlast);
  k_scan_carry<<<(B_ * D_ + 255) / 256, 256, 0, stream>>>(zlast, carryin);
  // 2) per-row stats, 3) gates
  k_stats<<<BT, 256, 0, stream>>>(x, kick_w, gate_w, stats);
  k_gates<<<(BT + 255) / 256, 256, 0, stream>>>(stats, gate_b, kick_b, g, kc);
  // 4) weight conversion + A-matrix build
  k_wcat<<<(int)((3ull * 1024 * 3072 + 255) / 256), 256, 0, stream>>>(Wp, Wi, Wd, Wcat);
  k_f32_to_bf16<<<(int)((1024ull * 1024 + 255) / 256), 256, 0, stream>>>(o_w, owb, 1024ull * 1024);
  k_buildA<<<(int)(((size_t)BT * D_ + 255) / 256), 256, 0, stream>>>(x, z, carryin, g, kc, kick_v, Abuf);
  // 5) qkv GEMMs (batch z=3): [4096x3072] x [1024x3072]^T -> bf16
  {
    dim3 grid(1024 / 128, BT / 128, 3);
    k_gemm_bf16<1><<<grid, 256, 0, stream>>>(Abuf, Wcat, nullptr, qkvb,
                                             BT, 1024, 3072,
                                             (long long)BT * 3072, 1024LL * 3072,
                                             (long long)BT * 1024);
  }
  // 6) dilated sparse attention (one wave32 per query)
  k_attn<<<(B_ * H_ * T_) / 8, 256, 0, stream>>>(qkvb, positions, attnb);
  // 7) output projection: [4096x1024] x [1024x1024]^T -> fp32 d_out
  {
    dim3 grid(1024 / 128, BT / 128, 1);
    k_gemm_bf16<0><<<grid, 256, 0, stream>>>(attnb, owb, outp, nullptr,
                                             BT, 1024, 1024, 0, 0, 0);
  }
}
